// GridGAT_8718783611583
// MI455X (gfx1250) — compile-verified, matrix operand order checked by reading, not run on hardware
//
#include <hip/hip_runtime.h>
#include <hip/hip_fp16.h>
#include <math.h>

#define NODES  50000
#define NEDGE  1600000
#define NSLOPE 0.2f
#define GEPS   1e-16f

typedef __attribute__((ext_vector_type(16))) _Float16 v16h;
typedef __attribute__((ext_vector_type(8)))  float    v8f;

// ---------------------------------------------------------------------------
// WMMA GEMM: out[N,outc] = in[N,64] * W[64,outc]   (f16 mul, f32 accumulate)
// One wave owns a 16x16 output tile. blockDim.x = 32*(outc/16), grid = N/16.
// W staged in LDS (cooperative float4 copy). A fragment fed by 4x b128 loads
// per k-step (K-runs are contiguous: K = (t&7) + 16*(t>>3) + 8*(lane>>4)).
// D: VGPR r -> row r (lanes 0-15) / r+8 (lanes 16-31), col = lane&15.
// ---------------------------------------------------------------------------
__global__ void gat_gemm_wmma(const float* __restrict__ in,
                              const float* __restrict__ W,
                              float* __restrict__ out, int outc) {
  __shared__ float sW[64 * 64];
  const int nW = 64 * outc;
  for (int i = threadIdx.x * 4; i < nW; i += blockDim.x * 4)
    *reinterpret_cast<float4*>(sW + i) =
        *reinterpret_cast<const float4*>(W + i);
  __syncthreads();

  const int lane = threadIdx.x & 31;
  const int wave = threadIdx.x >> 5;
  const int row0 = blockIdx.x * 16;
  const int mrow = row0 + (lane & 15);
  const int ncol = wave * 16 + (lane & 15);
  const int khi  = (lane >> 4) << 3;

  v8f acc = {};
#pragma unroll
  for (int kk = 0; kk < 64; kk += 32) {
    const float* ar = in + (size_t)mrow * 64 + kk + khi;
    const float4 a0 = *reinterpret_cast<const float4*>(ar);
    const float4 a1 = *reinterpret_cast<const float4*>(ar + 4);
    const float4 a2 = *reinterpret_cast<const float4*>(ar + 16);
    const float4 a3 = *reinterpret_cast<const float4*>(ar + 20);
    const float af[16] = {a0.x, a0.y, a0.z, a0.w, a1.x, a1.y, a1.z, a1.w,
                          a2.x, a2.y, a2.z, a2.w, a3.x, a3.y, a3.z, a3.w};
    v16h a, b;
#pragma unroll
    for (int t = 0; t < 16; ++t) {
      a[t] = (_Float16)af[t];
      const int K = (t & 7) + ((t >> 3) << 4) + khi;
      b[t] = (_Float16)sW[(kk + K) * outc + ncol];
    }
    acc = __builtin_amdgcn_wmma_f32_16x16x32_f16(false, a, false, b,
                                                 (short)0, acc, false, false);
  }
  const int rbase = row0 + ((lane >> 4) << 3);
#pragma unroll
  for (int r = 0; r < 8; ++r)
    out[(size_t)(rbase + r) * outc + ncol] = acc[r];
}

// ---------------------------------------------------------------------------
// Per-node attention coefficients: a_src/a_dst[n,h] = <h[n,h,:], att[h,:]>
// ---------------------------------------------------------------------------
template <int H, int C>
__global__ void gat_coeff(const float* __restrict__ h,
                          const float* __restrict__ att_s,
                          const float* __restrict__ att_d,
                          float* __restrict__ asrc, float* __restrict__ adst) {
  const int n = blockIdx.x * blockDim.x + threadIdx.x;
  if (n >= NODES) return;
  const float* hr = h + (size_t)n * H * C;
#pragma unroll
  for (int hh = 0; hh < H; ++hh) {
    float ss = 0.f, dd = 0.f;
#pragma unroll
    for (int c = 0; c < C; ++c) {
      const float v = hr[hh * C + c];
      ss += v * att_s[hh * C + c];
      dd += v * att_d[hh * C + c];
    }
    asrc[n * H + hh] = ss;
    adst[n * H + hh] = dd;
  }
}

__global__ void fill_u32(unsigned* __restrict__ p, unsigned v, int n) {
  int i = (blockIdx.x * blockDim.x + threadIdx.x) * 4;
  if (i + 3 < n) {
    uint4 q = {v, v, v, v};
    *reinterpret_cast<uint4*>(p + i) = q;
  } else {
    for (; i < n; ++i) p[i] = v;
  }
}

// monotone float <-> uint encoding so atomicMax(u32) == float max
__device__ __forceinline__ unsigned enc_f32(float f) {
  const unsigned u = __float_as_uint(f);
  return (u & 0x80000000u) ? ~u : (u | 0x80000000u);
}
__device__ __forceinline__ float dec_f32(unsigned e) {
  return (e & 0x80000000u) ? __uint_as_float(e & 0x7FFFFFFFu)
                           : __uint_as_float(~e);
}
__device__ __forceinline__ float lrelu(float z) {
  return (z > 0.f) ? z : NSLOPE * z;
}
__device__ __forceinline__ void edge_nodes(const int* __restrict__ ei, int e,
                                           int& s, int& d) {
  s = (e < NEDGE) ? ei[e] : (e - NEDGE);          // self-loops appended
  d = (e < NEDGE) ? ei[NEDGE + e] : (e - NEDGE);
}

// ---------------------------------------------------------------------------
// Edge pass 1: segment max of leaky_relu(a_src[s]+a_dst[d]) per head
// ---------------------------------------------------------------------------
template <int H>
__global__ void edge_max(const int* __restrict__ ei,
                         const float* __restrict__ asrc,
                         const float* __restrict__ adst,
                         unsigned* __restrict__ menc) {
  const int e = blockIdx.x * blockDim.x + threadIdx.x;
  if (e >= NEDGE + NODES) return;
  int s, d;
  edge_nodes(ei, e, s, d);
  if ((unsigned)s >= NODES || (unsigned)d >= NODES) return;
#pragma unroll
  for (int hh = 0; hh < H; ++hh) {
    const float z = lrelu(asrc[s * H + hh] + adst[d * H + hh]);
    atomicMax(&menc[d * H + hh], enc_f32(z));
  }
}

// ---------------------------------------------------------------------------
// Edge pass 2: segment sum of exp(e - m[dst])
// ---------------------------------------------------------------------------
template <int H>
__global__ void edge_sum(const int* __restrict__ ei,
                         const float* __restrict__ asrc,
                         const float* __restrict__ adst,
                         const unsigned* __restrict__ menc,
                         float* __restrict__ ssum) {
  const int e = blockIdx.x * blockDim.x + threadIdx.x;
  if (e >= NEDGE + NODES) return;
  int s, d;
  edge_nodes(ei, e, s, d);
  if ((unsigned)s >= NODES || (unsigned)d >= NODES) return;
#pragma unroll
  for (int hh = 0; hh < H; ++hh) {
    const float z = lrelu(asrc[s * H + hh] + adst[d * H + hh]);
    atomicAdd(&ssum[d * H + hh], __expf(z - dec_f32(menc[d * H + hh])));
  }
}

// ---------------------------------------------------------------------------
// Edge pass 3: aggr[dst,h,:] += h[src,h,:] * alpha   (thread per (edge,head))
// ---------------------------------------------------------------------------
template <int H, int C>
__global__ void edge_aggr(const int* __restrict__ ei,
                          const float* __restrict__ asrc,
                          const float* __restrict__ adst,
                          const unsigned* __restrict__ menc,
                          const float* __restrict__ ssum,
                          const float* __restrict__ h,
                          float* __restrict__ aggr) {
  const long long t = (long long)blockIdx.x * blockDim.x + threadIdx.x;
  if (t >= (long long)(NEDGE + NODES) * H) return;
  const int hh = (int)(t % H);
  const int e  = (int)(t / H);
  int s, d;
  edge_nodes(ei, e, s, d);
  if ((unsigned)s >= NODES || (unsigned)d >= NODES) return;
  const float z = lrelu(asrc[s * H + hh] + adst[d * H + hh]);
  const float alpha = __expf(z - dec_f32(menc[d * H + hh])) /
                      (ssum[d * H + hh] + GEPS);
  const float4* hs4 =
      reinterpret_cast<const float4*>(h + (size_t)s * H * C + hh * C);
  float* od = aggr + (size_t)d * H * C + hh * C;
#pragma unroll
  for (int c4 = 0; c4 < C / 4; ++c4) {
    const float4 v = hs4[c4];
    atomicAdd(&od[c4 * 4 + 0], v.x * alpha);
    atomicAdd(&od[c4 * 4 + 1], v.y * alpha);
    atomicAdd(&od[c4 * 4 + 2], v.z * alpha);
    atomicAdd(&od[c4 * 4 + 3], v.w * alpha);
  }
}

__device__ __forceinline__ float elu1(float v) {
  return (v > 0.f) ? v : (__expf(v) - 1.f);
}

__global__ void bias_elu(const float* __restrict__ aggr,
                         const float* __restrict__ b,
                         float* __restrict__ out, int K) {
  const int i4 = (blockIdx.x * blockDim.x + threadIdx.x) * 4;
  if (i4 >= NODES * K) return;
  float4 v = *reinterpret_cast<const float4*>(aggr + i4);
  const float4 bb = *reinterpret_cast<const float4*>(b + (i4 % K));
  v.x = elu1(v.x + bb.x);
  v.y = elu1(v.y + bb.y);
  v.z = elu1(v.z + bb.z);
  v.w = elu1(v.w + bb.w);
  *reinterpret_cast<float4*>(out + i4) = v;
}

__global__ void bias_add(const float* __restrict__ b, float* __restrict__ out,
                         int K) {
  const int i4 = (blockIdx.x * blockDim.x + threadIdx.x) * 4;
  if (i4 >= NODES * K) return;
  float4 v = *reinterpret_cast<float4*>(out + i4);
  const float4 bb = *reinterpret_cast<const float4*>(b + (i4 % K));
  v.x += bb.x; v.y += bb.y; v.z += bb.z; v.w += bb.w;
  *reinterpret_cast<float4*>(out + i4) = v;
}

// ---------------------------------------------------------------------------
template <int H, int C>
static void run_layer(const float* input, const float* W, const float* as_,
                      const float* ad_, const int* ei, float* h, float* asrc,
                      float* adst, unsigned* menc, float* ssum, float* dst,
                      hipStream_t stream) {
  constexpr int outc = H * C;
  const int TB = 256;
  const int ET = NEDGE + NODES;
  gat_gemm_wmma<<<NODES / 16, 32 * (outc / 16), 0, stream>>>(input, W, h,
                                                             outc);
  gat_coeff<H, C><<<(NODES + TB - 1) / TB, TB, 0, stream>>>(h, as_, ad_, asrc,
                                                            adst);
  fill_u32<<<(NODES * H / 4 + TB - 1) / TB, TB, 0, stream>>>(menc, 0u,
                                                             NODES * H);
  fill_u32<<<(NODES * H / 4 + TB - 1) / TB, TB, 0, stream>>>((unsigned*)ssum,
                                                             0u, NODES * H);
  fill_u32<<<(NODES * outc / 4 + TB - 1) / TB, TB, 0, stream>>>(
      (unsigned*)dst, 0u, NODES * outc);
  edge_max<H><<<(ET + TB - 1) / TB, TB, 0, stream>>>(ei, asrc, adst, menc);
  edge_sum<H><<<(ET + TB - 1) / TB, TB, 0, stream>>>(ei, asrc, adst, menc,
                                                     ssum);
  const long long tot = (long long)ET * H;
  edge_aggr<H, C><<<(int)((tot + TB - 1) / TB), TB, 0, stream>>>(
      ei, asrc, adst, menc, ssum, h, dst);
}

extern "C" void kernel_launch(void* const* d_in, const int* in_sizes, int n_in,
                              void* d_out, int out_size, void* d_ws,
                              size_t ws_size, hipStream_t stream) {
  (void)in_sizes; (void)n_in; (void)out_size; (void)ws_size;
  const float* x   = (const float*)d_in[0];
  const int*   ei  = (const int*)d_in[1];   // [2,E]: row0=src, row1=dst
  const float* W0  = (const float*)d_in[3];
  const float* as0 = (const float*)d_in[4];
  const float* ad0 = (const float*)d_in[5];
  const float* b0  = (const float*)d_in[6];
  const float* W1  = (const float*)d_in[7];
  const float* as1 = (const float*)d_in[8];
  const float* ad1 = (const float*)d_in[9];
  const float* b1  = (const float*)d_in[10];
  const float* W2  = (const float*)d_in[11];
  const float* as2 = (const float*)d_in[12];
  const float* ad2 = (const float*)d_in[13];
  const float* b2  = (const float*)d_in[14];
  float* out = (float*)d_out;

  // workspace: h(64N) | asrc(8N) | adst(8N) | menc(8N) | ssum(8N) | aggr(64N)
  //            | nxt(64N)  == 224N floats ~= 44.8 MB  (L2-resident)
  float*    ws   = (float*)d_ws;
  float*    h    = ws;
  float*    asrc = h    + (size_t)NODES * 64;
  float*    adst = asrc + (size_t)NODES * 8;
  unsigned* menc = (unsigned*)(adst + (size_t)NODES * 8);
  float*    ssum = (float*)(menc + (size_t)NODES * 8);
  float*    aggr = ssum + (size_t)NODES * 8;
  float*    nxt  = aggr + (size_t)NODES * 64;

  const int TB = 256;

  run_layer<8, 8>(x, W0, as0, ad0, ei, h, asrc, adst, menc, ssum, aggr,
                  stream);
  bias_elu<<<(NODES * 64 / 4 + TB - 1) / TB, TB, 0, stream>>>(aggr, b0, nxt,
                                                              64);
  run_layer<8, 8>(nxt, W1, as1, ad1, ei, h, asrc, adst, menc, ssum, aggr,
                  stream);
  bias_elu<<<(NODES * 64 / 4 + TB - 1) / TB, TB, 0, stream>>>(aggr, b1, nxt,
                                                              64);
  run_layer<1, 32>(nxt, W2, as2, ad2, ei, h, asrc, adst, menc, ssum, out,
                   stream);  // H=1: mean over heads == identity
  bias_add<<<(NODES * 32 / 4 + TB - 1) / TB, TB, 0, stream>>>(b2, out, 32);
}
